// GlobalLocalAttention_74887049773165
// MI455X (gfx1250) — matrix operand order repaired and checked by reading, not compiled, not run on hardware
//
#include <hip/hip_runtime.h>

// ---------------------------------------------------------------------------
// GlobalLocalAttention (contextual attention) for MI455X / gfx1250.
//
// Roofline: 2 GEMMs x 19.3 GFLOP x 2 samples = 77 GFLOP.  The 4096x4096
// attention matrices are kept in bf16 (67 MB each) so corr + scores together
// (134 MB) stay resident in the 192 MB L2 between kernels; HBM only sees the
// ~6 MB of real inputs/outputs.  All matmul work runs on
// v_wmma_f32_16x16x32_bf16 (bf16 in, f32 accumulate), and GEMM tile staging
// uses the CDNA5 async global->LDS copies (ASYNCcnt) instead of the
// VGPR round-trip.
// ---------------------------------------------------------------------------

typedef __attribute__((ext_vector_type(16))) __bf16 v16bf;
typedef __attribute__((ext_vector_type(8)))  __bf16 v8bf;
typedef __attribute__((ext_vector_type(8)))  float  v8f;

#define HW   64
#define LSZ  4096        // H*W
#define NC   64
#define KC   576         // NC * 9
#define EPSK 1e-7f
#define LDT  40          // LDS row stride (bf16 elems): 80 B, 16B-aligned, bank-skewed

// ---------------- CDNA5 async global->LDS copy (ASYNCcnt path) -------------
// the LDS destination address is taken from the vdst VGPR; generic shared pointers carry the LDS
// offset in addr[31:0] (ISA 10.2 aperture mapping), so truncation is the
// correct LDS byte address.
__device__ inline void async_copy16(void* lds, const void* gsrc) {
  const unsigned ldsa = (unsigned)(uintptr_t)lds;
  asm volatile("global_load_async_to_lds_b128 %0, %1, off"
               :: "v"(ldsa), "v"(gsrc)
               : "memory");
}
__device__ inline void wait_async0() {
  asm volatile("s_wait_asynccnt 0" ::: "memory");
}

// ---------------- WMMA fragment loads (per CDNA5 ISA 7.12.2 layouts) -------
// A-matrix 16x32 bf16: lane m<16 holds row m, K=0..7 (elems0-7), K=16..23
// (elems8-15); lane m+16 holds K=8..15 / K=24..31.
__device__ inline v16bf load_a_frag(const __bf16* tile, int lane) {
  const int r  = lane & 15;
  const int ko = (lane >> 4) * 8;
  const __bf16* p = tile + r * LDT;
  v8bf lo = *(const v8bf*)(p + ko);
  v8bf hi = *(const v8bf*)(p + ko + 16);
  return __builtin_shufflevector(lo, hi, 0,1,2,3,4,5,6,7,8,9,10,11,12,13,14,15);
}
// B-matrix 32x16 bf16: lane n<16 holds col n, K=0..7 / K=16..23; lane n+16
// holds K=8..15 / K=24..31 (same order as A).  Tile stored N-major (each LDS
// row = one col of B with K contiguous), so this is two 16B reads.
__device__ inline v16bf load_b_frag(const __bf16* tile, int lane) {
  const int c  = lane & 15;
  const int ko = (lane >> 4) * 8;
  const __bf16* p = tile + c * LDT;
  v8bf lo = *(const v8bf*)(p + ko);
  v8bf hi = *(const v8bf*)(p + ko + 16);
  return __builtin_shufflevector(lo, hi, 0,1,2,3,4,5,6,7,8,9,10,11,12,13,14,15);
}

// ---------------- prep: build bf16 operand matrices ------------------------
// Kmat [b][l][c*9+tap]       : L2-normalized bg patches (GEMM1 B operand)
// KmatT[b][tap][c][l]        : same values, transposed for GEMM2 A operand
// Fmat [b][p][c*9+tap]       : fg im2col patches        (GEMM1 A operand)
// mm   [b][l]                : 1.0 if 3x3 mask patch entirely zero
__global__ __launch_bounds__(64) void prep_kernel(
    const float* __restrict__ fg, const float* __restrict__ bg,
    const float* __restrict__ mask,
    __bf16* __restrict__ Kmat, __bf16* __restrict__ KmatT,
    __bf16* __restrict__ Fmat, float* __restrict__ mm)
{
  const int p  = blockIdx.x;
  const int bi = blockIdx.y;
  const int c  = threadIdx.x;            // channel 0..63
  const int y  = p >> 6, x = p & 63;

  __shared__ float sred[64];

  const float* fgb = fg + ((size_t)bi * NC + c) * LSZ;
  const float* bgb = bg + ((size_t)bi * NC + c) * LSZ;
  const float* mb  = mask + (size_t)bi * LSZ;

  float kv[9], fv[9];
  float ss = 0.f, msum = 0.f;
#pragma unroll
  for (int t = 0; t < 9; ++t) {
    const int yy = y + t / 3 - 1, xx = x + t % 3 - 1;
    const bool inb = ((unsigned)yy < HW) && ((unsigned)xx < HW);
    const float mv  = inb ? mb[yy * HW + xx] : 0.f;
    const float bv  = inb ? bgb[yy * HW + xx] * (1.f - mv) : 0.f;
    kv[t] = bv + EPSK;                  // reference adds EPS post-padding
    fv[t] = inb ? fgb[yy * HW + xx] : 0.f;
    ss   += kv[t] * kv[t];
    msum += mv;
  }
  sred[c] = ss;
  __syncthreads();
  for (int s = 32; s > 0; s >>= 1) {
    if (c < s) sred[c] += sred[c + s];
    __syncthreads();
  }
  const float rn = rsqrtf(sred[0]);     // norm > 0 thanks to EPS

  __bf16* krow = Kmat + ((size_t)bi * LSZ + p) * KC + c * 9;
  __bf16* frow = Fmat + ((size_t)bi * LSZ + p) * KC + c * 9;
#pragma unroll
  for (int t = 0; t < 9; ++t) {
    const __bf16 kvb = (__bf16)(kv[t] * rn);
    krow[t] = kvb;
    KmatT[(((size_t)bi * 9 + t) * NC + c) * LSZ + p] = kvb;
    frow[t] = (__bf16)fv[t];
  }
  if (c == 0) mm[(size_t)bi * LSZ + p] = (msum == 0.f) ? 1.f : 0.f;
}

// ---------------- GEMM1: corrT[p][l] = sum_k Fmat[p][k] * Kmat[l][k] -------
// M=4096 (p), N=4096 (l), K=576.  Block tile 128x64, 8 waves x (32x32).
#define G1_MBLK 128
#define G1_NBLK 64

__global__ __launch_bounds__(256) void gemm_corr_kernel(
    const __bf16* __restrict__ Fmat, const __bf16* __restrict__ Kmat,
    __bf16* __restrict__ corrT)
{
  __shared__ __align__(16) __bf16 As[G1_MBLK * LDT];
  __shared__ __align__(16) __bf16 Bs[G1_NBLK * LDT];

  const int bi    = blockIdx.z;
  const int pbase = blockIdx.x * G1_MBLK;
  const int lbase = blockIdx.y * G1_NBLK;
  const int tid   = threadIdx.x;
  const int lane  = tid & 31;
  const int wave  = tid >> 5;
  const int wm    = (wave & 3) * 32;     // 0,32,64,96
  const int wn    = (wave >> 2) * 32;    // 0,32

  const __bf16* Fb = Fmat + (size_t)bi * LSZ * KC;
  const __bf16* Kb = Kmat + (size_t)bi * LSZ * KC;

  v8f acc00 = {}, acc01 = {}, acc10 = {}, acc11 = {};

  const int ar = tid >> 1, ah = (tid & 1) * 16;   // A stage: 128 rows x 32
  const int br = tid >> 2, bq = (tid & 3) * 8;    // B stage:  64 rows x 32

  const __bf16* arow = Fb + (size_t)(pbase + ar) * KC + ah;
  const __bf16* brow = Kb + (size_t)(lbase + br) * KC + bq;
  __bf16* adst = As + ar * LDT + ah;
  __bf16* bdst = Bs + br * LDT + bq;

  for (int k0 = 0; k0 < KC; k0 += 32) {
    async_copy16(adst,     arow + k0);
    async_copy16(adst + 8, arow + k0 + 8);
    async_copy16(bdst,     brow + k0);
    if (k0 + 32 < KC) {
      __builtin_prefetch(arow + k0 + 32, 0, 1);
      __builtin_prefetch(brow + k0 + 32, 0, 1);
    }
    wait_async0();
    __syncthreads();
    const v16bf a0 = load_a_frag(As + (wm     ) * LDT, lane);
    const v16bf a1 = load_a_frag(As + (wm + 16) * LDT, lane);
    const v16bf b0 = load_b_frag(Bs + (wn     ) * LDT, lane);
    const v16bf b1 = load_b_frag(Bs + (wn + 16) * LDT, lane);
    acc00 = __builtin_amdgcn_wmma_f32_16x16x32_bf16(false, a0, false, b0, (short)0, acc00, false, false);
    acc01 = __builtin_amdgcn_wmma_f32_16x16x32_bf16(false, a0, false, b1, (short)0, acc01, false, false);
    acc10 = __builtin_amdgcn_wmma_f32_16x16x32_bf16(false, a1, false, b0, (short)0, acc10, false, false);
    acc11 = __builtin_amdgcn_wmma_f32_16x16x32_bf16(false, a1, false, b1, (short)0, acc11, false, false);
    __syncthreads();
  }

  // C/D layout: elem v -> (M = v + (lane>=16)*8, N = lane&15)
  const int nl = lane & 15;
  const int mo = (lane >> 4) * 8;
  __bf16* cb = corrT + (size_t)bi * LSZ * LSZ;
#pragma unroll
  for (int v = 0; v < 8; ++v) {
    cb[(size_t)(pbase + wm      + mo + v) * LSZ + (lbase + wn      + nl)] = (__bf16)acc00[v];
    cb[(size_t)(pbase + wm      + mo + v) * LSZ + (lbase + wn + 16 + nl)] = (__bf16)acc01[v];
    cb[(size_t)(pbase + wm + 16 + mo + v) * LSZ + (lbase + wn      + nl)] = (__bf16)acc10[v];
    cb[(size_t)(pbase + wm + 16 + mo + v) * LSZ + (lbase + wn + 16 + nl)] = (__bf16)acc11[v];
  }
}

// ---------------- softmax: 3x3 propagation + mm gate + softmax_l + argmax --
__global__ __launch_bounds__(256) void softmax_prop_kernel(
    const __bf16* __restrict__ corrT, const float* __restrict__ mm,
    __bf16* __restrict__ scoresT, float* __restrict__ out)
{
  const int p  = blockIdx.x;
  const int bi = blockIdx.y;
  const int y  = p >> 6, x = p & 63;
  const int tid = threadIdx.x;

  __shared__ float rf[256];
  __shared__ int   ri[256];

  const __bf16* cb = corrT + (size_t)bi * LSZ * LSZ;
  const float*  mb = mm + (size_t)bi * LSZ;

  const __bf16* rows[9];
  int nn = 0;
#pragma unroll
  for (int t = 0; t < 9; ++t) {
    const int yy = y + t / 3 - 1, xx = x + t % 3 - 1;
    if (((unsigned)yy < HW) && ((unsigned)xx < HW))
      rows[nn++] = cb + (size_t)(yy * HW + xx) * LSZ;
  }

  float lg[16], vm[16];
  float lmax = -1e30f;
  float bestv = -1e30f;
  int   besti = LSZ;                    // sentinel: "no valid l"
#pragma unroll
  for (int i = 0; i < 16; ++i) {
    const int l = tid + i * 256;
    float t = 0.f;
    for (int j = 0; j < nn; ++j) t += (float)rows[j][l];
    const float m   = mb[l];
    const float lgv = 10.f * t * m;     // gated logit (invalid -> 0, as ref)
    lg[i] = lgv; vm[i] = m;
    lmax = fmaxf(lmax, lgv);
    if (m > 0.f && (lgv > bestv)) { bestv = lgv; besti = l; }
  }

  // block max (softmax stabilization, matches jax softmax)
  rf[tid] = lmax; __syncthreads();
  for (int s = 128; s > 0; s >>= 1) {
    if (tid < s) rf[tid] = fmaxf(rf[tid], rf[tid + s]);
    __syncthreads();
  }
  const float gmax = rf[0]; __syncthreads();

  float ex[16], lsum = 0.f;
#pragma unroll
  for (int i = 0; i < 16; ++i) { ex[i] = __expf(lg[i] - gmax); lsum += ex[i]; }
  rf[tid] = lsum; __syncthreads();
  for (int s = 128; s > 0; s >>= 1) {
    if (tid < s) rf[tid] += rf[tid + s];
    __syncthreads();
  }
  const float inv = 1.f / rf[0]; __syncthreads();

  __bf16* srow = scoresT + ((size_t)bi * LSZ + p) * LSZ;
#pragma unroll
  for (int i = 0; i < 16; ++i)
    srow[tid + i * 256] = (__bf16)(ex[i] * inv * vm[i]);

  // argmax over valid l, first-index tie-break
  rf[tid] = bestv; ri[tid] = besti; __syncthreads();
  for (int s = 128; s > 0; s >>= 1) {
    if (tid < s) {
      const float ov = rf[tid + s]; const int oi = ri[tid + s];
      if (ov > rf[tid] || (ov == rf[tid] && oi < ri[tid])) { rf[tid] = ov; ri[tid] = oi; }
    }
    __syncthreads();
  }
  if (tid == 0) {
    const int idx = (ri[0] >= LSZ) ? 0 : ri[0];
    float* flow = out + (size_t)2 * NC * LSZ;       // after attended
    flow[((size_t)bi * 2 + 0) * LSZ + p] = (float)(idx >> 6) - (float)y;
    flow[((size_t)bi * 2 + 1) * LSZ + p] = (float)(idx & 63) - (float)x;
  }
}

// ---------------- GEMM2: rec + mask blend ----------------------------------
// rec[c][p] = sum_{tap,l} kernels[l][c][flip(tap)] * scores[l][p+shift(tap)]
// M=64 (c), N=128/block (p), K = 4096 per tap x 9 taps (accumulated).
// 8 waves x (32x32) tiles -> 32 WMMA per block per k-step (same sync density
// as GEMM1).
#define G2_NBLK 128

__global__ __launch_bounds__(256) void gemm_rec_kernel(
    const __bf16* __restrict__ KmatT, const __bf16* __restrict__ scoresT,
    const float* __restrict__ fg, const float* __restrict__ mask,
    float* __restrict__ out)
{
  __shared__ __align__(16) __bf16 As[NC * LDT];
  __shared__ __align__(16) __bf16 Bs[G2_NBLK * LDT];

  const int bi    = blockIdx.y;
  const int pbase = blockIdx.x * G2_NBLK;
  const int tid   = threadIdx.x;
  const int lane  = tid & 31;
  const int wave  = tid >> 5;
  const int wm    = (wave & 1) * 32;    // channel tile: 0 / 32
  const int wn    = (wave >> 1) * 32;   // p tile: 0/32/64/96

  const __bf16* Sb = scoresT + (size_t)bi * LSZ * LSZ;

  v8f acc00 = {}, acc01 = {}, acc10 = {}, acc11 = {};

  const int sr = tid >> 2, sq = (tid & 3) * 8;     // A stage:  64 rows x 32
  const int r2 = tid >> 1, h2 = (tid & 1) * 16;    // B stage: 128 rows x 32
  __bf16* adst = As + sr * LDT + sq;
  __bf16* bdst = Bs + r2 * LDT + h2;

  for (int tap = 0; tap < 9; ++tap) {
    const int dy = tap / 3 - 1, dx = tap % 3 - 1;
    const int tapf = 8 - tap;           // kernel flip: (2-kh,2-kw)
    const __bf16* arow =
        KmatT + ((size_t)bi * 9 + tapf) * NC * LSZ + (size_t)sr * LSZ + sq;

    const int pr = pbase + r2;
    const int yy = (pr >> 6) + dy, xx = (pr & 63) + dx;
    const bool vld = ((unsigned)yy < HW) && ((unsigned)xx < HW);
    const __bf16* brow = vld ? (Sb + (size_t)(yy * HW + xx) * LSZ + h2) : nullptr;
    const uint4 bz = make_uint4(0u, 0u, 0u, 0u);

    for (int l0 = 0; l0 < LSZ; l0 += 32) {
      async_copy16(adst, arow + l0);
      if (vld) {
        async_copy16(bdst,     brow + l0);
        async_copy16(bdst + 8, brow + l0 + 8);
      } else {
        *(uint4*)(bdst)     = bz;
        *(uint4*)(bdst + 8) = bz;
      }
      if (l0 + 32 < LSZ) {
        __builtin_prefetch(arow + l0 + 32, 0, 1);
        if (vld) __builtin_prefetch(brow + l0 + 32, 0, 1);
      }
      wait_async0();
      __syncthreads();
      const v16bf a0 = load_a_frag(As + (wm     ) * LDT, lane);
      const v16bf a1 = load_a_frag(As + (wm + 16) * LDT, lane);
      const v16bf b0 = load_b_frag(Bs + (wn     ) * LDT, lane);
      const v16bf b1 = load_b_frag(Bs + (wn + 16) * LDT, lane);
      acc00 = __builtin_amdgcn_wmma_f32_16x16x32_bf16(false, a0, false, b0, (short)0, acc00, false, false);
      acc01 = __builtin_amdgcn_wmma_f32_16x16x32_bf16(false, a0, false, b1, (short)0, acc01, false, false);
      acc10 = __builtin_amdgcn_wmma_f32_16x16x32_bf16(false, a1, false, b0, (short)0, acc10, false, false);
      acc11 = __builtin_amdgcn_wmma_f32_16x16x32_bf16(false, a1, false, b1, (short)0, acc11, false, false);
      __syncthreads();
    }
  }

  // epilogue: final = rec * m^2 / 9 + fg * (1 - m)
  const int nl = lane & 15;
  const int mo = (lane >> 4) * 8;
  const int pc0 = pbase + wn + nl;
  const int pc1 = pc0 + 16;
  const float mv0 = mask[(size_t)bi * LSZ + pc0];
  const float mv1 = mask[(size_t)bi * LSZ + pc1];
  const float s20 = mv0 * mv0 * (1.f / 9.f), bl0 = 1.f - mv0;
  const float s21 = mv1 * mv1 * (1.f / 9.f), bl1 = 1.f - mv1;
  const float* fgb = fg + (size_t)bi * NC * LSZ;
#pragma unroll
  for (int v = 0; v < 8; ++v) {
    const int c0 = wm + mo + v;
    const int c1 = wm + 16 + mo + v;
    out[((size_t)bi * NC + c0) * LSZ + pc0] =
        acc00[v] * s20 + fgb[(size_t)c0 * LSZ + pc0] * bl0;
    out[((size_t)bi * NC + c0) * LSZ + pc1] =
        acc01[v] * s21 + fgb[(size_t)c0 * LSZ + pc1] * bl1;
    out[((size_t)bi * NC + c1) * LSZ + pc0] =
        acc10[v] * s20 + fgb[(size_t)c1 * LSZ + pc0] * bl0;
    out[((size_t)bi * NC + c1) * LSZ + pc1] =
        acc11[v] * s21 + fgb[(size_t)c1 * LSZ + pc1] * bl1;
  }
}

// ---------------- launcher -------------------------------------------------
extern "C" void kernel_launch(void* const* d_in, const int* in_sizes, int n_in,
                              void* d_out, int out_size, void* d_ws, size_t ws_size,
                              hipStream_t stream) {
  (void)in_sizes; (void)n_in; (void)out_size; (void)ws_size;
  const float* fg   = (const float*)d_in[0];
  const float* bg   = (const float*)d_in[1];
  const float* mask = (const float*)d_in[2];
  float* out = (float*)d_out;

  char* ws = (char*)d_ws;
  size_t off = 0;
  auto take = [&](size_t bytes) -> char* {
    char* p = ws + off;
    off = (off + bytes + 255) & ~(size_t)255;
    return p;
  };
  __bf16* Kmat    = (__bf16*)take((size_t)2 * LSZ * KC * 2);      //  9.4 MB
  __bf16* KmatT   = (__bf16*)take((size_t)2 * 9 * NC * LSZ * 2);  //  9.4 MB
  __bf16* Fmat    = (__bf16*)take((size_t)2 * LSZ * KC * 2);      //  9.4 MB
  __bf16* corrT   = (__bf16*)take((size_t)2 * LSZ * LSZ * 2);     // 67.1 MB (L2-resident)
  __bf16* scoresT = (__bf16*)take((size_t)2 * LSZ * LSZ * 2);     // 67.1 MB (L2-resident)
  float*  mm      = (float*)take((size_t)2 * LSZ * 4);

  prep_kernel<<<dim3(LSZ, 2), 64, 0, stream>>>(fg, bg, mask, Kmat, KmatT, Fmat, mm);
  gemm_corr_kernel<<<dim3(LSZ / G1_MBLK, LSZ / G1_NBLK, 2), 256, 0, stream>>>(Fmat, Kmat, corrT);
  softmax_prop_kernel<<<dim3(LSZ, 2), 256, 0, stream>>>(corrT, mm, scoresT, out);
  gemm_rec_kernel<<<dim3(LSZ / G2_NBLK, 2), 256, 0, stream>>>(KmatT, scoresT, fg, mask, out);
}
